// GraphSAGE_Pointnet_70162585747537
// MI455X (gfx1250) — matrix operand order repaired and checked by reading, not compiled, not run on hardware
//
#include <hip/hip_runtime.h>
#include <hip/hip_bf16.h>
#include <math.h>

typedef __attribute__((ext_vector_type(16))) _Float16 v16h;
typedef __attribute__((ext_vector_type(8)))  float    v8f;

#define NB 4
#define NP 4096
#define KN 32
#define CH 64
#define CNT_BNK (NB*NP*KN)   /* 524288 rows for per-neighbor BNs */
#define CNT_BN  (NB*NP)      /* 16384 rows for per-point BNs     */
#define PTS_PER_BLOCK 4

#define WMMA_F16(A, Bm, C) \
  __builtin_amdgcn_wmma_f32_16x16x32_f16(false, (A), false, (Bm), (short)0, (C), false, false)

// K index inside a 32-wide chunk for A-matrix slot j (contiguous: k = 16*half + j, matches B fills)
__device__ __forceinline__ int a_kidx(int j, int half) {
  return half * 16 + j;
}

__device__ __forceinline__ float bn_apply(float v, float s, float s2, float inv_cnt,
                                          float g, float bb) {
  float m   = s * inv_cnt;
  float var = s2 * inv_cnt - m * m;
  return (v - m) * rsqrtf(var + 1e-5f) * g + bb;
}

// ---------------------------------------------------------------- stats zero
__global__ void zero_stats_kernel(float* __restrict__ stats) {
  int i = blockIdx.x * 256 + threadIdx.x;
  if (i < 640) stats[i] = 0.0f;
}

// ---------------------------------------------------------------- kNN (top-32)
__global__ void knn_kernel(const float* __restrict__ x, int* __restrict__ idxw) {
  __shared__ float xs0[NP], xs1[NP], xs2[NP];
  const int b     = blockIdx.x >> 4;          // 16 blocks per batch
  const int nbase = (blockIdx.x & 15) * 256;
  const int tid   = threadIdx.x;
  const float* xb = x + (size_t)b * 3 * NP;
  for (int i = tid; i < NP; i += 256) {
    xs0[i] = xb[i];
    xs1[i] = xb[NP + i];
    xs2[i] = xb[2 * NP + i];
  }
  __syncthreads();
  const int n = nbase + tid;
  const float cx = xs0[n], cy = xs1[n], cz = xs2[n];
  const float sqn = cx * cx + cy * cy + cz * cz;
  float bd[KN];
  int   bi[KN];
#pragma unroll
  for (int j = 0; j < KN; ++j) { bd[j] = 3.4e38f; bi[j] = 0; }
  for (int m = 0; m < NP; ++m) {
    float mx = xs0[m], my = xs1[m], mz = xs2[m];
    float sqm = mx * mx + my * my + mz * mz;
    float d = sqn + sqm - 2.0f * (cx * mx + cy * my + cz * mz);
    if (d < bd[KN - 1]) {
      int p = 0;
#pragma unroll
      for (int j = 0; j < KN; ++j) p += (bd[j] <= d) ? 1 : 0;
#pragma unroll
      for (int j = KN - 1; j >= 1; --j) {
        if (j > p)       { bd[j] = bd[j - 1]; bi[j] = bi[j - 1]; }
        else if (j == p) { bd[j] = d;         bi[j] = m;         }
      }
      if (p == 0) { bd[0] = d; bi[0] = m; }
    }
  }
  int* op = idxw + (size_t)(b * NP + n) * KN;
#pragma unroll
  for (int j = 0; j < KN; ++j) op[j] = bi[j];
}

// ------------------------------------------------- BN stats for t1 = rel@Wp1
__global__ void statsp_kernel(const float* __restrict__ x, const int* __restrict__ idxw,
                              const float* __restrict__ Wp1, float* __restrict__ stats) {
  __shared__ float ss[128];
  const int tid = threadIdx.x;                 // 256
  if (tid < 128) ss[tid] = 0.0f;
  __syncthreads();
  const int ch = tid & 63;
  const int rs = tid >> 6;                     // 0..3
  const float w0 = Wp1[ch * 3 + 0], w1 = Wp1[ch * 3 + 1], w2 = Wp1[ch * 3 + 2];
  float s = 0.0f, s2 = 0.0f;
  const int base = blockIdx.x * 1024;
  for (int it = 0; it < 256; ++it) {
    int row = base + it * 4 + rs;
    __builtin_prefetch(&idxw[row + 64], 0, 0);   // gfx1250 global_prefetch
    int b   = row >> 17;                       // / (NP*KN)
    int rem = row & (NP * KN - 1);
    int n   = rem >> 5;
    int id  = idxw[row];
    const float* xb = x + (size_t)b * 3 * NP;
    float r0 = xb[id]          - xb[n];
    float r1 = xb[NP + id]     - xb[NP + n];
    float r2 = xb[2 * NP + id] - xb[2 * NP + n];
    float v = r0 * w0 + r1 * w1 + r2 * w2;
    s += v; s2 += v * v;
  }
  atomicAdd(&ss[ch], s);
  atomicAdd(&ss[64 + ch], s2);
  __syncthreads();
  if (tid < 128) atomicAdd(&stats[tid], ss[tid]);
}

// ------------- fused per-point attention (one block = PTS_PER_BLOCK points, 64 thr)
// Weights staged into LDS once per block and reused across the points.
template <bool STATS_ONLY>
__global__ void attn_kernel(const float* __restrict__ x,  const int* __restrict__ idxw,
                            const float* __restrict__ Wq, const float* __restrict__ Wkv,
                            const float* __restrict__ Wp1,const float* __restrict__ Wp2,
                            const float* __restrict__ Wa, const float* __restrict__ Wsk,
                            const float* __restrict__ g_p,const float* __restrict__ b_p,
                            const float* __restrict__ g_a,const float* __restrict__ b_a,
                            float* __restrict__ stats,
                            float* __restrict__ aggw, float* __restrict__ skpre) {
  __shared__ _Float16 Wp2h[CH * CH];
  __shared__ _Float16 Wah[CH * CH];
  __shared__ float Wp1s[CH * 3];
  __shared__ float Wqs[CH * 3];
  __shared__ float Wkvs[128 * 3];
  __shared__ float Wsks[CH * 6];
  __shared__ float REL[KN][3];
  __shared__ float Qs[CH];
  __shared__ _Float16 Uh[KN][CH];
  __shared__ _Float16 A2h[KN][CH];
  __shared__ float POS[KN][CH];
  __shared__ float VF[KN][CH];
  __shared__ float LG[KN][CH];
  __shared__ float sstat[128];
  __shared__ float nbm[3];

  const int tid = threadIdx.x;                 // 0..63
  const float inv_bnk = 1.0f / (float)CNT_BNK;

  // ---- stage all weights once per block
  for (int i = tid; i < CH * CH; i += 64) { Wp2h[i] = (_Float16)Wp2[i]; Wah[i] = (_Float16)Wa[i]; }
  for (int i = tid; i < CH * 3; i += 64)  { Wp1s[i] = Wp1[i]; Wqs[i] = Wq[i]; }
  for (int i = tid; i < 128 * 3; i += 64) Wkvs[i] = Wkv[i];
  for (int i = tid; i < CH * 6; i += 64)  Wsks[i] = Wsk[i];
  for (int i = tid; i < 128; i += 64)     sstat[i] = 0.0f;
  __syncthreads();

  const int w = tid >> 5, lane = tid & 31, half = lane >> 4, mrow = lane & 15;
  const int rowBase = w * 16;

  for (int it = 0; it < PTS_PER_BLOCK; ++it) {
    const int p = blockIdx.x * PTS_PER_BLOCK + it;
    const int b = p >> 12;
    const int n = p & (NP - 1);
    const float* xb = x + (size_t)b * 3 * NP;
    const float c0 = xb[n], c1 = xb[NP + n], c2 = xb[2 * NP + n];

    if (tid < KN) {
      int id = idxw[(size_t)p * KN + tid];
      REL[tid][0] = xb[id]          - c0;
      REL[tid][1] = xb[NP + id]     - c1;
      REL[tid][2] = xb[2 * NP + id] - c2;
    }
    __syncthreads();

    Qs[tid] = c0 * Wqs[tid * 3] + c1 * Wqs[tid * 3 + 1] + c2 * Wqs[tid * 3 + 2];

    // t1 = rel@Wp1 -> BN_p -> relu -> Uh (f16)
    {
      int row = tid & 31, cb = (tid >> 5) * 32;
      float r0 = REL[row][0], r1 = REL[row][1], r2 = REL[row][2];
      for (int ch = cb; ch < cb + 32; ++ch) {
        float t = r0 * Wp1s[ch * 3] + r1 * Wp1s[ch * 3 + 1] + r2 * Wp1s[ch * 3 + 2];
        float u = bn_apply(t, stats[ch], stats[64 + ch], inv_bnk, g_p[ch], b_p[ch]);
        Uh[row][ch] = (_Float16)fmaxf(u, 0.0f);
      }
    }
    __syncthreads();

    // pos = relu(BN(t1)) @ Wp2^T  via WMMA (K=64 -> 2 steps, 4 col tiles)
    {
      v16h a0, a1;
#pragma unroll
      for (int j = 0; j < 16; ++j) {
        a0[j] = Uh[rowBase + mrow][a_kidx(j, half)];
        a1[j] = Uh[rowBase + mrow][32 + a_kidx(j, half)];
      }
#pragma unroll
      for (int tile = 0; tile < 4; ++tile) {
        int e = tile * 16 + mrow;
        v16h b0, b1;
#pragma unroll
        for (int j = 0; j < 16; ++j) {
          b0[j] = Wp2h[e * 64 + half * 16 + j];
          b1[j] = Wp2h[e * 64 + 32 + half * 16 + j];
        }
        v8f acc = {};
        acc = WMMA_F16(a0, b0, acc);
        acc = WMMA_F16(a1, b1, acc);
#pragma unroll
        for (int r = 0; r < 8; ++r)
          POS[rowBase + r + 8 * half][tile * 16 + mrow] = acc[r];
      }
    }
    __syncthreads();

    // kf, vf, A2 = q - kf + pos (f16)
    {
      int row = tid & 31, cb = (tid >> 5) * 32;
      float r0 = REL[row][0], r1 = REL[row][1], r2 = REL[row][2];
      for (int ch = cb; ch < cb + 32; ++ch) {
        float kf = r0 * Wkvs[ch * 3] + r1 * Wkvs[ch * 3 + 1] + r2 * Wkvs[ch * 3 + 2];
        int cv = 64 + ch;
        float vf = r0 * Wkvs[cv * 3] + r1 * Wkvs[cv * 3 + 1] + r2 * Wkvs[cv * 3 + 2];
        VF[row][ch]  = vf;
        A2h[row][ch] = (_Float16)(Qs[ch] - kf + POS[row][ch]);
      }
    }
    __syncthreads();

    // logits_pre = A2 @ Wa^T via WMMA
    {
      v16h a0, a1;
#pragma unroll
      for (int j = 0; j < 16; ++j) {
        a0[j] = A2h[rowBase + mrow][a_kidx(j, half)];
        a1[j] = A2h[rowBase + mrow][32 + a_kidx(j, half)];
      }
#pragma unroll
      for (int tile = 0; tile < 4; ++tile) {
        int e = tile * 16 + mrow;
        v16h b0, b1;
#pragma unroll
        for (int j = 0; j < 16; ++j) {
          b0[j] = Wah[e * 64 + half * 16 + j];
          b1[j] = Wah[e * 64 + 32 + half * 16 + j];
        }
        v8f acc = {};
        acc = WMMA_F16(a0, b0, acc);
        acc = WMMA_F16(a1, b1, acc);
#pragma unroll
        for (int r = 0; r < 8; ++r) {
          float v = acc[r];
          int ch  = tile * 16 + mrow;
          if constexpr (STATS_ONLY) {
            atomicAdd(&sstat[ch], v);
            atomicAdd(&sstat[64 + ch], v * v);
          } else {
            int kk = rowBase + r + 8 * half;
            float lg = bn_apply(v, stats[128 + ch], stats[192 + ch], inv_bnk,
                                g_a[ch], b_a[ch]);
            LG[kk][ch] = fmaxf(lg, 0.0f);
          }
        }
      }
    }

    if constexpr (!STATS_ONLY) {
      __syncthreads();
      // softmax over k per channel, then agg = sum attn*(vf+pos)
      {
        int ch = tid;
        float mx = -3.4e38f;
        for (int kk = 0; kk < KN; ++kk) mx = fmaxf(mx, LG[kk][ch]);
        float s = 0.0f;
        for (int kk = 0; kk < KN; ++kk) s += __expf(LG[kk][ch] - mx);
        float inv = 1.0f / s;
        float ag = 0.0f;
        for (int kk = 0; kk < KN; ++kk)
          ag += __expf(LG[kk][ch] - mx) * inv * (VF[kk][ch] + POS[kk][ch]);
        aggw[(size_t)p * CH + ch] = ag;
      }
      if (tid < 3) {
        float c = (tid == 0) ? c0 : ((tid == 1) ? c1 : c2);
        float m = -3.4e38f;
        for (int kk = 0; kk < KN; ++kk) m = fmaxf(m, REL[kk][tid] + c);
        nbm[tid] = m;
      }
      __syncthreads();
      // sk_pre = concat(x, nb_max) @ Wsk^T   (+ BN_sk stats)
      {
        int d = tid;
        float skv = c0 * Wsks[d * 6]     + c1 * Wsks[d * 6 + 1] + c2 * Wsks[d * 6 + 2]
                  + nbm[0] * Wsks[d * 6 + 3] + nbm[1] * Wsks[d * 6 + 4] + nbm[2] * Wsks[d * 6 + 5];
        skpre[(size_t)p * CH + d] = skv;
        atomicAdd(&stats[384 + d], skv);
        atomicAdd(&stats[448 + d], skv * skv);
      }
    }
    __syncthreads();   // protect shared buffers before next point's writes
  }

  if constexpr (STATS_ONLY) {
    for (int i = tid; i < 128; i += 64) atomicAdd(&stats[128 + i], sstat[i]);
  }
}

// ---------------------------------------------- h_pre = agg @ Wc1^T (WMMA)
__global__ void hpre_kernel(const float* __restrict__ agg, const float* __restrict__ Wc1,
                            float* __restrict__ hpre, float* __restrict__ stats) {
  __shared__ float ss[128];
  const int tid = threadIdx.x;                 // 128
  if (tid < 128) ss[tid] = 0.0f;
  __syncthreads();
  const int w = tid >> 5, lane = tid & 31, half = lane >> 4, mrow = lane & 15;
  const int rowBase = (blockIdx.x * 4 + w) * 16;
  v16h a0, a1;
#pragma unroll
  for (int j = 0; j < 16; ++j) {
    a0[j] = (_Float16)agg[(size_t)(rowBase + mrow) * CH + a_kidx(j, half)];
    a1[j] = (_Float16)agg[(size_t)(rowBase + mrow) * CH + 32 + a_kidx(j, half)];
  }
#pragma unroll
  for (int tile = 0; tile < 4; ++tile) {
    int e = tile * 16 + mrow;
    v16h b0, b1;
#pragma unroll
    for (int j = 0; j < 16; ++j) {
      b0[j] = (_Float16)Wc1[e * 64 + half * 16 + j];
      b1[j] = (_Float16)Wc1[e * 64 + 32 + half * 16 + j];
    }
    v8f acc = {};
    acc = WMMA_F16(a0, b0, acc);
    acc = WMMA_F16(a1, b1, acc);
#pragma unroll
    for (int r = 0; r < 8; ++r) {
      float v = acc[r];
      hpre[(size_t)(rowBase + r + 8 * half) * CH + tile * 16 + mrow] = v;
      atomicAdd(&ss[tile * 16 + mrow], v);
      atomicAdd(&ss[64 + tile * 16 + mrow], v * v);
    }
  }
  __syncthreads();
  if (tid < 128) atomicAdd(&stats[256 + tid], ss[tid]);
}

// ------------------- out_pre = concat(relu(BN(h)), relu(BN(sk))) @ Wga^T (K=128)
__global__ void outpre_kernel(const float* __restrict__ hpre, const float* __restrict__ skpre,
                              const float* __restrict__ Wga,
                              const float* __restrict__ g_c,  const float* __restrict__ b_c,
                              const float* __restrict__ g_sk, const float* __restrict__ b_sk,
                              float* __restrict__ outpre, float* __restrict__ stats) {
  __shared__ float ss[128];
  const int tid = threadIdx.x;                 // 128
  if (tid < 128) ss[tid] = 0.0f;
  __syncthreads();
  const float inv_bn = 1.0f / (float)CNT_BN;
  const int w = tid >> 5, lane = tid & 31, half = lane >> 4, mrow = lane & 15;
  const int rowBase = (blockIdx.x * 4 + w) * 16;
  const size_t rowOff = (size_t)(rowBase + mrow) * CH;
  v16h aK[4];
#pragma unroll
  for (int ks = 0; ks < 4; ++ks) {
#pragma unroll
    for (int j = 0; j < 16; ++j) {
      int ch = (ks & 1) * 32 + a_kidx(j, half);
      float v;
      if (ks < 2) {
        float raw = hpre[rowOff + ch];
        v = fmaxf(bn_apply(raw, stats[256 + ch], stats[320 + ch], inv_bn, g_c[ch], b_c[ch]), 0.0f);
      } else {
        float raw = skpre[rowOff + ch];
        v = fmaxf(bn_apply(raw, stats[384 + ch], stats[448 + ch], inv_bn, g_sk[ch], b_sk[ch]), 0.0f);
      }
      aK[ks][j] = (_Float16)v;
    }
  }
#pragma unroll
  for (int tile = 0; tile < 4; ++tile) {
    int e = tile * 16 + mrow;
    v8f acc = {};
#pragma unroll
    for (int ks = 0; ks < 4; ++ks) {
      v16h bv;
#pragma unroll
      for (int j = 0; j < 16; ++j)
        bv[j] = (_Float16)Wga[e * 128 + ks * 32 + half * 16 + j];
      acc = WMMA_F16(aK[ks], bv, acc);
    }
#pragma unroll
    for (int r = 0; r < 8; ++r) {
      float v = acc[r];
      outpre[(size_t)(rowBase + r + 8 * half) * CH + tile * 16 + mrow] = v;
      atomicAdd(&ss[tile * 16 + mrow], v);
      atomicAdd(&ss[64 + tile * 16 + mrow], v * v);
    }
  }
  __syncthreads();
  if (tid < 128) atomicAdd(&stats[512 + tid], ss[tid]);
}

// -------------------------------- out = relu(BN_ga(out_pre)) transposed to (B,64,N)
__global__ void finalize_kernel(const float* __restrict__ outpre,
                                const float* __restrict__ g_ga, const float* __restrict__ b_ga,
                                const float* __restrict__ stats, float* __restrict__ out) {
  int i = blockIdx.x * 256 + threadIdx.x;      // i = (b*64+e)*NP + n
  int b = i >> 18;
  int e = (i >> 12) & 63;
  int n = i & (NP - 1);
  const float inv_bn = 1.0f / (float)CNT_BN;
  float raw = outpre[(size_t)((b << 12) + n) * CH + e];
  out[i] = fmaxf(bn_apply(raw, stats[512 + e], stats[576 + e], inv_bn, g_ga[e], b_ga[e]), 0.0f);
}

extern "C" void kernel_launch(void* const* d_in, const int* in_sizes, int n_in,
                              void* d_out, int out_size, void* d_ws, size_t ws_size,
                              hipStream_t stream) {
  (void)in_sizes; (void)n_in; (void)out_size; (void)ws_size;
  const float* x    = (const float*)d_in[0];
  const float* Wq   = (const float*)d_in[1];
  const float* Wkv  = (const float*)d_in[2];
  const float* Wp1  = (const float*)d_in[3];
  const float* Wp2  = (const float*)d_in[4];
  const float* Wa   = (const float*)d_in[5];
  const float* Wc1  = (const float*)d_in[6];
  const float* Wsk  = (const float*)d_in[7];
  const float* Wga  = (const float*)d_in[8];
  const float* g_p  = (const float*)d_in[9];
  const float* b_p  = (const float*)d_in[10];
  const float* g_a  = (const float*)d_in[11];
  const float* b_a  = (const float*)d_in[12];
  const float* g_c  = (const float*)d_in[13];
  const float* b_c  = (const float*)d_in[14];
  const float* g_sk = (const float*)d_in[15];
  const float* b_sk = (const float*)d_in[16];
  const float* g_ga = (const float*)d_in[17];
  const float* b_ga = (const float*)d_in[18];

  // workspace layout: stats(640 f) | idx(B*N*K i32) | agg | h_pre | sk_pre | out_pre
  float* stats  = (float*)d_ws;
  int*   idxw   = (int*)((char*)d_ws + 2560);
  float* aggw   = (float*)((char*)d_ws + 2560 + (size_t)CNT_BNK * sizeof(int));
  float* hpre   = aggw  + (size_t)CNT_BN * CH;
  float* skpre  = hpre  + (size_t)CNT_BN * CH;
  float* outpre = skpre + (size_t)CNT_BN * CH;
  float* out    = (float*)d_out;

  zero_stats_kernel<<<3, 256, 0, stream>>>(stats);
  knn_kernel<<<64, 256, 0, stream>>>(x, idxw);
  statsp_kernel<<<512, 256, 0, stream>>>(x, idxw, Wp1, stats);
  attn_kernel<true><<<CNT_BN / PTS_PER_BLOCK, 64, 0, stream>>>(
      x, idxw, Wq, Wkv, Wp1, Wp2, Wa, Wsk, g_p, b_p, g_a, b_a, stats, aggw, skpre);
  attn_kernel<false><<<CNT_BN / PTS_PER_BLOCK, 64, 0, stream>>>(
      x, idxw, Wq, Wkv, Wp1, Wp2, Wa, Wsk, g_p, b_p, g_a, b_a, stats, aggw, skpre);
  hpre_kernel<<<CNT_BN / 64, 128, 0, stream>>>(aggw, Wc1, hpre, stats);
  outpre_kernel<<<CNT_BN / 64, 128, 0, stream>>>(hpre, skpre, Wga, g_c, b_c, g_sk, b_sk,
                                                 outpre, stats);
  finalize_kernel<<<(NB * CH * NP) / 256, 256, 0, stream>>>(outpre, g_ga, b_ga, stats, out);
}
